// EdgeGNN_layer_48962627174424
// MI455X (gfx1250) — matrix operand order repaired and checked by reading, not compiled, not run on hardware
//
#include <hip/hip_runtime.h>
#include <hip/hip_bf16.h>

// ---------------------------------------------------------------------------
// EdgeGNN layer for MI455X (gfx1250, wave32, WMMA).
// Dense GEMMs run as bf16 WMMA (f32 accumulate); SpMM is L2-resident f32
// atomic scatter; GRU gates fused with the final two GEMMs.
//
// v2: accumulator chains seeded from a single shared zero v8f (kills the
// bias-splat v_dual_mov chains); biases folded into the post-GEMM
// elementwise stage; B-fragment loads hoisted per K-step so the compiler
// clauses them and overlaps them with WMMA issue.
// ---------------------------------------------------------------------------

typedef __attribute__((ext_vector_type(16))) __bf16    v16bf;
typedef __attribute__((ext_vector_type(8)))  float     v8f;
typedef __attribute__((ext_vector_type(8)))  unsigned  v8u;

#define WMMA_BF16(A, B, C) \
  __builtin_amdgcn_wmma_f32_16x16x32_bf16(false, (A), false, (B), (short)0, (C), false, false)

#define WAIT_DS() asm volatile("s_wait_dscnt 0" ::: "memory")

__device__ __forceinline__ unsigned pack_bf16(float a, float b) {
  unsigned ua = __float_as_uint(a), ub = __float_as_uint(b);
  unsigned ra = (ua + 0x7FFFu + ((ua >> 16) & 1u)) >> 16;
  unsigned rb = (ub + 0x7FFFu + ((ub >> 16) & 1u)) >> 16;
  return (ra & 0xFFFFu) | (rb << 16);
}

__device__ __forceinline__ unsigned short bf16_of(float a) {
  unsigned ua = __float_as_uint(a);
  return (unsigned short)((ua + 0x7FFFu + ((ua >> 16) & 1u)) >> 16);
}

__device__ __forceinline__ float sigmoidf_(float x) { return 1.f / (1.f + __expf(-x)); }

// A-fragment (16x32 bf16) from an LDS tile (row-major bf16, even stride).
// ISA layout: lanes 0-15 row M=lane, lanes 16-31 row M=lane-16;
// VGPR v<4 -> K = 8*half + 2v, v>=4 -> K = 16 + 8*half + 2(v-4)  (K pairs).
__device__ __forceinline__ v16bf load_a_frag(const unsigned short* tile, int strideElems,
                                             int kbase, int lane) {
  const int mrow = lane & 15, half = lane >> 4;
  union { v16bf v; unsigned u[8]; } f;
  const unsigned* row32 = (const unsigned*)(tile + mrow * strideElems);
#pragma unroll
  for (int v = 0; v < 8; ++v) {
    int k0 = (v < 4) ? (8 * half + 2 * v) : (16 + 8 * half + 2 * (v - 4));
    f.u[v] = row32[(kbase + k0) >> 1];
  }
  return f.v;
}

// B-fragment: pre-packed in workspace as [frag][lane][8 dwords] (1 KB/fragment).
__device__ __forceinline__ v16bf load_b_frag(const unsigned* __restrict__ fragBuf,
                                             int fragIdx, int lane) {
  union { v16bf v; v8u u; } f;
  f.u = *(const v8u*)(fragBuf + (size_t)fragIdx * 256 + lane * 8);
  return f.v;
}

// Stage a 16x128 f32 global tile into a 16x136 bf16 LDS tile.
__device__ __forceinline__ void stage_tile16x128(unsigned* t32, const float* __restrict__ src,
                                                 int lane) {
#pragma unroll
  for (int i = 0; i < 32; ++i) {
    int p = i * 32 + lane;            // 1024 dword-pairs total
    int row = p >> 6, cp = p & 63;
    float2 v = ((const float2*)(src + row * 128))[cp];
    t32[row * 68 + cp] = pack_bf16(v.x, v.y);
  }
}

// ---------------------------------------------------------------------------
// Kernel 0: pack fp32 weights [N x K] into bf16 WMMA B-fragments.
// B layout (32x16 bf16): lane n = l&15; half = l>>4; VGPR v holds
// {B[k0][n], B[k0 + 1][n]}, k0 = v<4 ? 8*half + 2v : 16 + 8*half + 2(v-4), where B[k][n] = W[n][k].
// ---------------------------------------------------------------------------
__global__ void pack_weights(const float* __restrict__ W, int N, int K,
                             unsigned* __restrict__ out, int ntiles, int ktiles) {
  int idx = blockIdx.x * blockDim.x + threadIdx.x;
  int total = ntiles * ktiles * 256;
  if (idx >= total) return;
  int v    = idx & 7;
  int lane = (idx >> 3) & 31;
  int tile = idx >> 8;
  int kt = tile % ktiles, nt = tile / ktiles;
  int n  = nt * 16 + (lane & 15);
  int k0 = kt * 32 + ((v < 4) ? (8 * (lane >> 4) + 2 * v) : (16 + 8 * (lane >> 4) + 2 * (v - 4)));
  float a = (n < N && k0     < K) ? W[(size_t)n * K + k0    ] : 0.f;
  float b = (n < N && k0 + 1 < K) ? W[(size_t)n * K + k0 + 1] : 0.f;
  out[idx] = pack_bf16(a, b);
}

__global__ void zero_f32(float* __restrict__ p, size_t n) {
  size_t i = (size_t)blockIdx.x * blockDim.x + threadIdx.x;
  size_t stride = (size_t)gridDim.x * blockDim.x;
  for (; i < n; i += stride) p[i] = 0.f;
}

// ---------------------------------------------------------------------------
// Kernel 1: msg1 = relu(concat(message_old, edge_feat) @ W1^T + b1)   [E,128]
// K = 144 zero-padded to 160 (5 K-tiles of 32).
// ---------------------------------------------------------------------------
__global__ __launch_bounds__(128) void gemm1_relu(
    const float* __restrict__ mo, const float* __restrict__ ef,
    const unsigned* __restrict__ w1f, const float* __restrict__ b1,
    float* __restrict__ msg1, int E) {
  __shared__ unsigned short lds[4][16 * 168];
  const int lane = threadIdx.x & 31;
  const int wave = threadIdx.x >> 5;
  const int r0 = (blockIdx.x * 4 + wave) * 16;
  if (r0 >= E) return;
  unsigned* t32 = (unsigned*)lds[wave];
  const int ln = lane & 15, half = lane >> 4;

  // Stage concat tile: 16 rows x 84 dwords (168 bf16), zero-pad K 144..159.
  for (int row = 0; row < 16; ++row) {
    const float* mrow = mo + (size_t)(r0 + row) * 128;
    const float* erow = ef + (size_t)(r0 + row) * 16;
#pragma unroll
    for (int i = 0; i < 3; ++i) {
      int cp = i * 32 + lane;
      if (cp < 84) {
        int col = cp * 2;
        unsigned pv = 0u;
        if (col < 128)      { float2 v = *(const float2*)(mrow + col);        pv = pack_bf16(v.x, v.y); }
        else if (col < 144) { float2 v = *(const float2*)(erow + col - 128);  pv = pack_bf16(v.x, v.y); }
        t32[row * 84 + cp] = pv;
      }
    }
  }
  WAIT_DS();

  v16bf a[5];
#pragma unroll
  for (int kt = 0; kt < 5; ++kt) a[kt] = load_a_frag((unsigned short*)t32, 168, kt * 32, lane);

  const v8f vzero = {};
#pragma unroll
  for (int nt = 0; nt < 8; ++nt) {
    // Peel kt=0 with the shared zero C operand (no accumulator-init movs).
    v8f c = WMMA_BF16(a[0], load_b_frag(w1f, nt * 5 + 0, lane), vzero);
#pragma unroll
    for (int kt = 1; kt < 5; ++kt) c = WMMA_BF16(a[kt], load_b_frag(w1f, nt * 5 + kt, lane), c);
    float bias = b1[nt * 16 + ln];
#pragma unroll
    for (int r = 0; r < 8; ++r)
      msg1[(size_t)(r0 + half * 8 + r) * 128 + nt * 16 + ln] = fmaxf(c[r] + bias, 0.f);
  }
}

// ---------------------------------------------------------------------------
// Kernel 2: acc[r] += val * msg1[c]  (one wave per nnz, float4 per lane).
// msg1/acc are L2-resident (102 MB each, 192 MB L2).
// ---------------------------------------------------------------------------
__global__ __launch_bounds__(256) void spmm_atomic(
    const float* __restrict__ msg1, const int* __restrict__ rows,
    const int* __restrict__ cols, const float* __restrict__ vals,
    float* __restrict__ acc, int nnz) {
  int wid  = (blockIdx.x * blockDim.x + threadIdx.x) >> 5;
  int lane = threadIdx.x & 31;
  int nw   = (gridDim.x * blockDim.x) >> 5;
  for (int i = wid; i < nnz; i += nw) {
    int r = rows[i], c = cols[i];
    float v = vals[i];
    float4 m = ((const float4*)(msg1 + (size_t)c * 128))[lane];
    float* dst = acc + (size_t)r * 128 + lane * 4;
    atomicAdd(dst + 0, m.x * v);
    atomicAdd(dst + 1, m.y * v);
    atomicAdd(dst + 2, m.z * v);
    atomicAdd(dst + 3, m.w * v);
  }
}

// ---------------------------------------------------------------------------
// Kernel 3: msg2 = relu(acc @ W2^T + b2);  GRU(msg2, message_old) -> out.
// Per wave: 16 rows; 32 WMMA for GEMM2, 24 WMMA per output N-tile for gates.
// ---------------------------------------------------------------------------
__global__ __launch_bounds__(128) void fused_gru(
    const float* __restrict__ acc, const float* __restrict__ mo,
    const unsigned* __restrict__ w2f, const unsigned* __restrict__ wihf,
    const unsigned* __restrict__ whhf,
    const float* __restrict__ b2, const float* __restrict__ bih,
    const float* __restrict__ bhh,
    float* __restrict__ out, int E) {
  __shared__ unsigned short lds[4][16 * 136];
  const int lane = threadIdx.x & 31;
  const int wave = threadIdx.x >> 5;
  const int r0 = (blockIdx.x * 4 + wave) * 16;
  if (r0 >= E) return;
  unsigned short* tile = lds[wave];
  unsigned* t32 = (unsigned*)tile;
  const int ln = lane & 15, half = lane >> 4;
  const v8f vzero = {};

  // ---- GEMM2: msg2 = relu(acc @ W2^T + b2), staged through LDS ----
  stage_tile16x128(t32, acc + (size_t)r0 * 128, lane);
  WAIT_DS();
  v16bf aA[4];
#pragma unroll
  for (int kt = 0; kt < 4; ++kt) aA[kt] = load_a_frag(tile, 136, kt * 32, lane);

#pragma unroll
  for (int nt = 0; nt < 8; ++nt) {
    v8f c = WMMA_BF16(aA[0], load_b_frag(w2f, nt * 4 + 0, lane), vzero);
#pragma unroll
    for (int kt = 1; kt < 4; ++kt) c = WMMA_BF16(aA[kt], load_b_frag(w2f, nt * 4 + kt, lane), c);
    float bias = b2[nt * 16 + ln];
#pragma unroll
    for (int r = 0; r < 8; ++r)  // C-frag row = r + 8*half, col = nt*16 + ln
      tile[(half * 8 + r) * 136 + nt * 16 + ln] = bf16_of(fmaxf(c[r] + bias, 0.f));
  }
  WAIT_DS();
  v16bf aM[4];  // msg2 as A operand
#pragma unroll
  for (int kt = 0; kt < 4; ++kt) aM[kt] = load_a_frag(tile, 136, kt * 32, lane);

  // ---- message_old as A operand ----
  stage_tile16x128(t32, mo + (size_t)r0 * 128, lane);
  WAIT_DS();
  v16bf aH[4];
#pragma unroll
  for (int kt = 0; kt < 4; ++kt) aH[kt] = load_a_frag(tile, 136, kt * 32, lane);

  // ---- GRU gates, one output N-tile at a time (24 WMMA each) ----
  const float* moRow = mo + (size_t)r0 * 128;
#pragma unroll 1
  for (int nt = 0; nt < 8; ++nt) {
    const int n = nt * 16 + ln;

    // kt = 0 peeled: seed all six accumulator chains from the shared zero.
    v16bf fIR = load_b_frag(wihf, (nt)      * 4 + 0, lane);
    v16bf fHR = load_b_frag(whhf, (nt)      * 4 + 0, lane);
    v16bf fIZ = load_b_frag(wihf, (8 + nt)  * 4 + 0, lane);
    v16bf fHZ = load_b_frag(whhf, (8 + nt)  * 4 + 0, lane);
    v16bf fIN = load_b_frag(wihf, (16 + nt) * 4 + 0, lane);
    v16bf fHN = load_b_frag(whhf, (16 + nt) * 4 + 0, lane);
    v8f ir  = WMMA_BF16(aM[0], fIR, vzero);
    v8f hr  = WMMA_BF16(aH[0], fHR, vzero);
    v8f iz  = WMMA_BF16(aM[0], fIZ, vzero);
    v8f hz  = WMMA_BF16(aH[0], fHZ, vzero);
    v8f inn = WMMA_BF16(aM[0], fIN, vzero);
    v8f hn  = WMMA_BF16(aH[0], fHN, vzero);

#pragma unroll
    for (int kt = 1; kt < 4; ++kt) {
      // Hoist the six B-fragment loads so they clause and overlap the WMMAs.
      v16bf gIR = load_b_frag(wihf, (nt)      * 4 + kt, lane);
      v16bf gHR = load_b_frag(whhf, (nt)      * 4 + kt, lane);
      v16bf gIZ = load_b_frag(wihf, (8 + nt)  * 4 + kt, lane);
      v16bf gHZ = load_b_frag(whhf, (8 + nt)  * 4 + kt, lane);
      v16bf gIN = load_b_frag(wihf, (16 + nt) * 4 + kt, lane);
      v16bf gHN = load_b_frag(whhf, (16 + nt) * 4 + kt, lane);
      ir  = WMMA_BF16(aM[kt], gIR, ir);
      hr  = WMMA_BF16(aH[kt], gHR, hr);
      iz  = WMMA_BF16(aM[kt], gIZ, iz);
      hz  = WMMA_BF16(aH[kt], gHZ, hz);
      inn = WMMA_BF16(aM[kt], gIN, inn);
      hn  = WMMA_BF16(aH[kt], gHN, hn);
    }

    // Bias pairs fold into the gate pre-activations.
    const float br = bih[n]       + bhh[n];
    const float bz = bih[128 + n] + bhh[128 + n];
    const float bi = bih[256 + n];
    const float bh = bhh[256 + n];
#pragma unroll
    for (int r = 0; r < 8; ++r) {
      int rowg = half * 8 + r;
      float rr = sigmoidf_(ir[r] + hr[r] + br);
      float zz = sigmoidf_(iz[r] + hz[r] + bz);
      float nn = tanhf((inn[r] + bi) + rr * (hn[r] + bh));
      float h  = moRow[rowg * 128 + n];
      out[(size_t)(r0 + rowg) * 128 + n] = (1.f - zz) * nn + zz * h;
    }
  }
}

// ---------------------------------------------------------------------------
extern "C" void kernel_launch(void* const* d_in, const int* in_sizes, int n_in,
                              void* d_out, int out_size, void* d_ws, size_t ws_size,
                              hipStream_t stream) {
  (void)n_in; (void)out_size; (void)ws_size;
  const float* edge_feat   = (const float*)d_in[0];
  const float* message_old = (const float*)d_in[1];
  const int*   adj_rows    = (const int*)d_in[2];
  const int*   adj_cols    = (const int*)d_in[3];
  const float* adj_vals    = (const float*)d_in[4];
  const float* W1   = (const float*)d_in[5];
  const float* b1   = (const float*)d_in[6];
  const float* W2   = (const float*)d_in[7];
  const float* b2   = (const float*)d_in[8];
  const float* W_ih = (const float*)d_in[9];
  const float* W_hh = (const float*)d_in[10];
  const float* b_ih = (const float*)d_in[11];
  const float* b_hh = (const float*)d_in[12];

  const int E   = in_sizes[0] / 16;  // edge_feat is [E, 16]
  const int NNZ = in_sizes[2];

  // Workspace layout: msg1 (E*128 f32) | acc (E*128 f32) | bf16 weight frags.
  float*    msg1 = (float*)d_ws;
  float*    acc  = msg1 + (size_t)E * 128;
  unsigned* w1f  = (unsigned*)(acc + (size_t)E * 128);  // 8*5 frags
  unsigned* w2f  = w1f  + 8 * 5 * 256;                  // 8*4 frags
  unsigned* wihf = w2f  + 8 * 4 * 256;                  // 24*4 frags
  unsigned* whhf = wihf + 24 * 4 * 256;                 // 24*4 frags

  pack_weights<<<40, 256, 0, stream>>>(W1,   128, 144, w1f,  8,  5);
  pack_weights<<<32, 256, 0, stream>>>(W2,   128, 128, w2f,  8,  4);
  pack_weights<<<96, 256, 0, stream>>>(W_ih, 384, 128, wihf, 24, 4);
  pack_weights<<<96, 256, 0, stream>>>(W_hh, 384, 128, whhf, 24, 4);

  zero_f32<<<2048, 256, 0, stream>>>(acc, (size_t)E * 128);

  const int tileBlocks = (E + 63) / 64;  // 4 waves/block, 16 rows/wave
  gemm1_relu<<<tileBlocks, 128, 0, stream>>>(message_old, edge_feat, w1f, b1, msg1, E);
  spmm_atomic<<<2048, 256, 0, stream>>>(msg1, adj_rows, adj_cols, adj_vals, acc, NNZ);
  fused_gru<<<tileBlocks, 128, 0, stream>>>(acc, message_old, w2f, wihf, whhf,
                                            b2, b_ih, b_hh, (float*)d_out, E);
}
